// FFB_encoder_515396075700
// MI455X (gfx1250) — matrix operand order repaired and checked by reading, not compiled, not run on hardware
//
#include <hip/hip_runtime.h>
#include <math.h>

typedef __attribute__((ext_vector_type(16))) _Float16 v16h;
typedef __attribute__((ext_vector_type(8)))  _Float16 v8h;
typedef __attribute__((ext_vector_type(8)))  float    v8f;
typedef __attribute__((ext_vector_type(4)))  unsigned int u32x4;
typedef __attribute__((ext_vector_type(8)))  int      i32x8;
typedef __attribute__((ext_vector_type(4)))  int      i32x4;

#define N_PTS   131072
#define G_LVL   5
#define F_DIM   8
#define W_DIM   256
#define OUT_DIM 64
#define M_TILE  128
#define SIN_W0f 5.0f
#define TWO_PIf 6.28318530717958647692f

// ---------------- workspace layout ----------------
// Per-layer contiguous blob, ALREADY granule-swizzled so the staged LDS image
// is a pure linear copy (TDM-friendly):
//   [Wh_swz 256rows*512B | Whh_swz 64rows*512B | ffn_swz 256rows*64B]
constexpr size_t LAYER_BYTES = 131072 + 32768 + 16384;   // 180224
constexpr size_t LAYER_HALFS = LAYER_BYTES / 2;          // 90112
constexpr size_t WHS_OFF_H   = 0;                        // 65536 halfs
constexpr size_t WHHS_OFF_H  = 65536;                    // 16384 halfs
constexpr size_t FFNS_OFF_H  = 81920;                    // 8192 halfs

// ---------------- dynamic LDS layout (bytes) ----------------
// xs : 8 waves * 16 rows * 512B (row = 256 f16, 16B-granule rotated)   65536
// weights blob (TDM dest, byte-identical to workspace blob)           180224
#define XS_OFF   0
#define WS_OFF   65536
#define WSH_OFF  (WS_OFF + 131072)
#define WF_OFF   (WSH_OFF + 32768)
#define LDS_BYTES (WS_OFF + (int)LAYER_BYTES)   // 245760 <= 320KB/WGP

// ---------------- helpers ----------------
__device__ __forceinline__ v8f zero8f() {
    v8f r;
#pragma unroll
    for (int i = 0; i < 8; ++i) r[i] = 0.0f;
    return r;
}
__device__ __forceinline__ v16h zero16h() {
    v16h r;
#pragma unroll
    for (int i = 0; i < 16; ++i) r[i] = (_Float16)0.0f;
    return r;
}
__device__ __forceinline__ v16h cat16(v8h lo, v8h hi) {
    v16h r;
#pragma unroll
    for (int i = 0; i < 8; ++i) { r[i] = lo[i]; r[i + 8] = hi[i]; }
    return r;
}
__device__ __forceinline__ v8f wmma_f16(v16h a, v16h b, v8f c) {
    return __builtin_amdgcn_wmma_f32_16x16x32_f16(false, a, false, b, (short)0, c, false, false);
}

// xs strip address: per-wave private, row = 512B, 16B granule rotation vs row
__device__ __forceinline__ int xs_addr(int wave, int row, int col) {
    int gran = ((col >> 3) + ((row & 7) << 2)) & 31;
    return XS_OFF + wave * 8192 + row * 512 + gran * 16 + (col & 7) * 2;
}

// A fragment (16x32 f16): lane<16 -> row=lane, K=k0..k0+7 & k0+16..23
//                         lane>=16 -> row=lane-16, K=k0+8..15 & k0+24..31
__device__ __forceinline__ v16h load_a_frag(const char* sm, int wave, int lane, int kb) {
    const int row = lane & 15;
    const int c0  = kb * 32 + ((lane >> 4) << 3);
    v8h lo = *(const v8h*)(sm + xs_addr(wave, row, c0));
    v8h hi = *(const v8h*)(sm + xs_addr(wave, row, c0 + 16));
    return cat16(lo, hi);
}

// B fragment (32x16 f16) from 512B-row swizzled LDS:
// lane<16 -> col n, K=kb*32+0..7 & +16..23 ; lane>=16 -> K=+8..15 & +24..31 (k0 includes +16)
__device__ __forceinline__ v16h load_b_frag512(const char* base, int n, int k0) {
    const int rot = (n & 7) << 2;
    const int g0  = ((k0 >> 5) << 2) + ((k0 >> 4) & 1);
    const char* row = base + n * 512;
    v8h lo = *(const v8h*)(row + (((g0    ) + rot) & 31) * 16);
    v8h hi = *(const v8h*)(row + (((g0 + 2) + rot) & 31) * 16);
    return cat16(lo, hi);
}
// B fragment from 64B-row swizzled LDS (ffn, K padded to 32)
__device__ __forceinline__ v16h load_b_frag64(const char* base, int n, int k0) {
    const int rot = n & 3;
    const int g0  = k0 >> 3;
    const char* row = base + n * 64;
    v8h lo = *(const v8h*)(row + (((g0    ) + rot) & 3) * 16);
    v8h hi = *(const v8h*)(row + (((g0 + 1) + rot) & 3) * 16);
    return cat16(lo, hi);
}

// ---- TDM: one descriptor copies the whole per-layer weight blob -> LDS ----
// D# group0: count=1 | lds_addr | global_addr[56:0] | type=2
// D# group1: data_size=3 (8B units), 1-D tile: tensor_dim0=tile_dim0=22528,
//            tensor_dim1=tile_dim1=1, tensor_dim0_stride=22528
__device__ __forceinline__ void tdm_stage_weights(const _Float16* gsrc, unsigned lds_byte_off) {
    constexpr unsigned elems = (unsigned)(LAYER_BYTES / 8);      // 22528 x 8B
    const unsigned long long ga = (unsigned long long)(size_t)gsrc;
    u32x4 g0;
    g0[0] = 1u;                                                  // count=1 (user)
    g0[1] = lds_byte_off;                                        // lds_addr
    g0[2] = (unsigned)ga;                                        // global_addr[31:0]
    g0[3] = (unsigned)((ga >> 32) & 0x01ffffffu) | (2u << 30);   // addr[56:32] | type=2
    i32x8 g1;
    g1[0] = (int)(3u << 16);                                     // wg_mask=0, data_size=8B
    g1[1] = (int)((elems & 0xffffu) << 16);                      // tensor_dim0[15:0]
    g1[2] = (int)(((elems >> 16) & 0xffffu) | (1u << 16));       // dim0[31:16] | tensor_dim1=1
    g1[3] = (int)((elems & 0xffffu) << 16);                      // dim1 hi=0 | tile_dim0
    g1[4] = 1;                                                   // tile_dim1=1, tile_dim2=0
    g1[5] = (int)elems;                                          // tensor_dim0_stride[31:0]
    g1[6] = 0;
    g1[7] = 0;
    const i32x4 gz4 = {0, 0, 0, 0};
    const i32x8 gz8 = {0, 0, 0, 0, 0, 0, 0, 0};
    // 6-arg toolchain form: (g0, g1, g2, g3, extra_group, cpol)
    __builtin_amdgcn_tensor_load_to_lds(g0, g1, gz4, gz4, gz8, 0);
}

// ---------------- prep: fold + transpose + f16-cast + pre-swizzle ----------------
__global__ __launch_bounds__(256) void prep_weights(
    const float* __restrict__ Wh, const float* __restrict__ Whh,
    const float* __restrict__ ffnA, _Float16* __restrict__ wblob)
{
    const int idx = blockIdx.x * 256 + threadIdx.x;

    if (idx < G_LVL * W_DIM * W_DIM) {               // Wh^T * 5, swizzled
        const int l = idx / (W_DIM * W_DIM);
        const int rem = idx - l * W_DIM * W_DIM;
        const int n = rem >> 8, k = rem & 255;
        const int swz = ((((k >> 3) + ((n & 7) << 2)) & 31) << 3) + (k & 7);
        wblob[(size_t)l * LAYER_HALFS + WHS_OFF_H + n * 256 + swz] =
            (_Float16)(SIN_W0f * Wh[((size_t)l * W_DIM + k) * W_DIM + n]);
    }
    if (idx < G_LVL * OUT_DIM * W_DIM) {             // Wh_high^T * 5, swizzled
        const int l = idx / (OUT_DIM * W_DIM);
        const int rem = idx - l * OUT_DIM * W_DIM;
        const int n = rem >> 8, k = rem & 255;
        const int swz = ((((k >> 3) + ((n & 7) << 2)) & 31) << 3) + (k & 7);
        wblob[(size_t)l * LAYER_HALFS + WHHS_OFF_H + n * 256 + swz] =
            (_Float16)(SIN_W0f * Whh[((size_t)l * W_DIM + k) * OUT_DIM + n]);
    }
    if (idx < G_LVL * W_DIM * 32) {                  // 2pi*2^l*ffn^T, K-padded, swizzled
        const int l = idx / (W_DIM * 32);
        const int rem = idx - l * W_DIM * 32;
        const int n = rem >> 5, k = rem & 31;
        float v = 0.0f;
        if (k < F_DIM) v = TWO_PIf * (float)(1 << l) * ffnA[((size_t)l * F_DIM + k) * W_DIM + n];
        const int swz = ((((k >> 3) + (n & 3)) & 3) << 3) + (k & 7);
        wblob[(size_t)l * LAYER_HALFS + FFNS_OFF_H + n * 32 + swz] = (_Float16)v;
    }
}

// ---------------- fully fused encoder ----------------
__global__ __launch_bounds__(256) void ffb_fused(
    const float* __restrict__ in_pos, const float* __restrict__ grid_feats,
    const float* __restrict__ W0, const float* __restrict__ b0,
    const float* __restrict__ bh, const float* __restrict__ bhh,
    const _Float16* __restrict__ wblob,
    float* __restrict__ out)
{
    extern __shared__ char sm[];
    const int tid  = threadIdx.x;
    const int wave = tid >> 5;
    const int lane = tid & 31;
    const int hi16 = lane >> 4;            // upper-half-wave flag
    const int nl   = lane & 15;
    const int rowBase = blockIdx.x * M_TILE + wave * 16;
    // low 32 bits of a generic LDS pointer == LDS byte offset (ISA 10.2 aperture map)
    const unsigned ldsWeightsOff = (unsigned)(size_t)sm + WS_OFF;

    // ---- layer 0: x = sin(5*(p@W0 + b0)) into private xs strip ----
    for (int rr = 0; rr < 16; ++rr) {
        const int gr = rowBase + rr;
        const float p0 = in_pos[gr * 3 + 0];
        const float p1 = in_pos[gr * 3 + 1];
        const float p2 = in_pos[gr * 3 + 2];
#pragma unroll
        for (int cc = 0; cc < 8; ++cc) {
            const int col = lane + (cc << 5);
            const float h = SIN_W0f * (p0 * W0[col] + p1 * W0[W_DIM + col] +
                                       p2 * W0[2 * W_DIM + col] + b0[col]);
            *(_Float16*)(sm + xs_addr(wave, rr, col)) = (_Float16)__sinf(h);
        }
    }

    // A fragments (16x256 f16 strip) held in registers
    v16h afr[8];
#pragma unroll
    for (int kb = 0; kb < 8; ++kb) afr[kb] = load_a_frag(sm, wave, lane, kb);

    v8f xout[4];
#pragma unroll
    for (int jo = 0; jo < 4; ++jo) xout[jo] = zero8f();

    for (int l = 0; l < G_LVL; ++l) {
        // ---- TDM weight staging: wave 0 drives the DMA engine ----
        __syncthreads();                               // everyone done with old weights
        if (wave == 0) {
            tdm_stage_weights(wblob + (size_t)l * LAYER_HALFS, ldsWeightsOff);
            __builtin_amdgcn_s_wait_tensorcnt(0);      // TENSORcnt -> 0
        }
        __syncthreads();                               // staged weights visible

        if (l + 1 < G_LVL)                             // warm GL2 for next layer's blob
            __builtin_prefetch(wblob + (size_t)(l + 1) * LAYER_HALFS + tid * 64, 0, 1);

        // ---- grid-branch A fragment: K=0..7 valid, rest zero ----
        v16h ga = zero16h();
        if (hi16 == 0) {
            const float4* gp = (const float4*)(grid_feats +
                               ((size_t)(rowBase + nl) * (G_LVL * F_DIM) + l * F_DIM));
            const float4 g0 = gp[0], g1 = gp[1];
            ga[0] = (_Float16)g0.x; ga[1] = (_Float16)g0.y;
            ga[2] = (_Float16)g0.z; ga[3] = (_Float16)g0.w;
            ga[4] = (_Float16)g1.x; ga[5] = (_Float16)g1.y;
            ga[6] = (_Float16)g1.z; ga[7] = (_Float16)g1.w;
        }

        // ---- newX = sin(gx) + sin(X@Wh + 5*bh), written back to xs strip ----
        for (int j = 0; j < 16; ++j) {
            v8f acc = zero8f();
#pragma unroll
            for (int kb = 0; kb < 8; ++kb) {
                v16h b = load_b_frag512(sm + WS_OFF, (j << 4) + nl,
                                        kb * 32 + (hi16 << 4));
                acc = wmma_f16(afr[kb], b, acc);
            }
            v16h bg = load_b_frag64(sm + WF_OFF, (j << 4) + nl, hi16 << 4);
            v8f gacc = wmma_f16(ga, bg, zero8f());

            const int col = (j << 4) + nl;
            const float bb = SIN_W0f * bh[l * W_DIM + col];
#pragma unroll
            for (int i = 0; i < 8; ++i) {
                const float nx = __sinf(gacc[i]) + __sinf(acc[i] + bb);
                *(_Float16*)(sm + xs_addr(wave, i + (hi16 << 3), col)) = (_Float16)nx;
            }
        }

        // ---- relayout: reload A fragments of newX ----
#pragma unroll
        for (int kb = 0; kb < 8; ++kb) afr[kb] = load_a_frag(sm, wave, lane, kb);

        // ---- x_out += sin(newX @ Wh_high + 5*bhh) ----
#pragma unroll
        for (int jo = 0; jo < 4; ++jo) {
            v8f hacc = zero8f();
#pragma unroll
            for (int kb = 0; kb < 8; ++kb) {
                v16h b = load_b_frag512(sm + WSH_OFF, (jo << 4) + nl,
                                        kb * 32 + (hi16 << 4));
                hacc = wmma_f16(afr[kb], b, hacc);
            }
            const int col = (jo << 4) + nl;
            const float bb = SIN_W0f * bhh[l * OUT_DIM + col];
#pragma unroll
            for (int i = 0; i < 8; ++i) xout[jo][i] += __sinf(hacc[i] + bb);
        }
    }

    // ---- write x_out [N,64] f32 ----
#pragma unroll
    for (int jo = 0; jo < 4; ++jo) {
#pragma unroll
        for (int i = 0; i < 8; ++i) {
            const int row = rowBase + i + (hi16 << 3);
            out[(size_t)row * OUT_DIM + (jo << 4) + nl] = xout[jo][i];
        }
    }
}

extern "C" void kernel_launch(void* const* d_in, const int* in_sizes, int n_in,
                              void* d_out, int out_size, void* d_ws, size_t ws_size,
                              hipStream_t stream) {
    (void)in_sizes; (void)n_in; (void)out_size; (void)ws_size;
    const float* in_pos   = (const float*)d_in[0];
    const float* grid_f   = (const float*)d_in[1];
    const float* ffn_A    = (const float*)d_in[2];
    const float* W0       = (const float*)d_in[3];
    const float* b0       = (const float*)d_in[4];
    const float* Wh       = (const float*)d_in[5];
    const float* bh       = (const float*)d_in[6];
    const float* Wh_high  = (const float*)d_in[7];
    const float* bh_high  = (const float*)d_in[8];
    float* out = (float*)d_out;

    _Float16* wblob = (_Float16*)d_ws;   // 5 * 180224 B = 901120 B

    const int prep_blocks = (G_LVL * W_DIM * W_DIM + 255) / 256;   // 1280
    prep_weights<<<prep_blocks, 256, 0, stream>>>(Wh, Wh_high, ffn_A, wblob);

    ffb_fused<<<dim3(N_PTS / M_TILE), 256, LDS_BYTES, stream>>>(
        in_pos, grid_f, W0, b0, bh, bh_high, wblob, out);
}
